// GroupingModule_19902878450132
// MI455X (gfx1250) — matrix operand order repaired and checked by reading, not compiled, not run on hardware
//
#include <hip/hip_runtime.h>
#include <hip/hip_bf16.h>

typedef __bf16 bf16_t;
typedef __attribute__((ext_vector_type(16))) __bf16 v16bf;
typedef __attribute__((ext_vector_type(8)))  float  v8f;

#define B_     8
#define N_     2048
#define NP_    2050      // padded rows per batch (one zero row each side)
#define D_     512
#define H_     1024
#define KTOT   1536
#define NCHUNK 48        // KTOT / 32
#define ROWDW  (D_ / 2)  // 256 dwords per padded-x row

// ---------------- f32 -> bf16 (round to nearest even) ----------------
static __device__ __forceinline__ unsigned short f32_to_bf16(float f) {
    union { float f; unsigned u; } v; v.f = f;
    unsigned u = v.u;
    unsigned r = u + 0x7FFFu + ((u >> 16) & 1u);
    return (unsigned short)(r >> 16);
}

// Convert x (B,N,D) f32 -> packed bf16 pairs into padded layout (B, N+2, D).
__global__ __launch_bounds__(256) void cvt_x_kernel(const float* __restrict__ x,
                                                    unsigned* __restrict__ xb) {
    int i = blockIdx.x * 256 + threadIdx.x;          // dword index, 4,194,304 total
    float a = x[2 * i];
    float b = x[2 * i + 1];
    int bt  = i >> 19;                               // / (N_*ROWDW) = 2^19
    int rem = i & ((1 << 19) - 1);
    xb[(size_t)bt * (NP_ * ROWDW) + ROWDW + rem] =
        (unsigned)f32_to_bf16(a) | ((unsigned)f32_to_bf16(b) << 16);
}

// Zero the padding rows (row 0 and row NP_-1 of each batch).
__global__ __launch_bounds__(256) void zero_pad_kernel(unsigned* __restrict__ xb) {
    int b = blockIdx.x;
    int t = threadIdx.x;                             // 256 threads = 1 row of dwords
    size_t base = (size_t)b * (NP_ * ROWDW);
    xb[base + t] = 0u;
    xb[base + (size_t)(NP_ - 1) * ROWDW + t] = 0u;
}

// Convert conv_w (H,D,3) f32 into WMMA-B-fragment-ready bf16 layout:
// bfrag[c][o][v]: c = K-chunk of 32 (t=c/16, dbase=(c%16)*32), o = out channel,
// v=0..7 -> K pairs (0,1)..(14,15); v=8..15 -> pairs (16,17)..(30,31); lanes 0-15 take v 0-3,8-11.
__global__ __launch_bounds__(256) void cvt_w_kernel(const float* __restrict__ w,
                                                    unsigned* __restrict__ bfrag) {
    int i = blockIdx.x * 256 + threadIdx.x;          // 786,432 dwords total
    int v = i & 15;
    int o = (i >> 4) & (H_ - 1);
    int c = i >> 14;
    int t = c >> 4;
    int dbase = (c & 15) << 5;
    int kk = (v < 8) ? (2 * v) : (16 + 2 * (v - 8));
    int d0 = dbase + kk;
    float a = w[o * KTOT + d0 * 3 + t];
    float b = w[o * KTOT + (d0 + 1) * 3 + t];
    bfrag[i] = (unsigned)f32_to_bf16(a) | ((unsigned)f32_to_bf16(b) << 16);
}

// --------------------- fragment helpers ---------------------
union BFU { uint4 u[2]; v16bf v; };

static __device__ __forceinline__ void load_a(BFU& a, const unsigned* __restrict__ abase,
                                              int c) {
    const int t = c >> 4;
    const int db = (c & 15) << 4;                    // dword offset within row
    const uint4* ap = (const uint4*)(abase + t * ROWDW + db);
    a.u[0] = ap[0];                                  // K {0..7}  (+half*8)
    a.u[1] = ap[2];                                  // K {16..23}(+half*8)
}

static __device__ __forceinline__ void load_b(BFU* __restrict__ bb,
                                              const unsigned* __restrict__ bfrag,
                                              int c, int colbase, int half) {
    #pragma unroll
    for (int j = 0; j < 8; ++j) {
        const uint4* bp = (const uint4*)(bfrag +
            (((size_t)c * H_ + colbase + (j << 4)) << 4) + (half << 2));
        bb[j].u[0] = bp[0];
        bb[j].u[1] = bp[2];
    }
}

static __device__ __forceinline__ void mma8(v8f* __restrict__ acc, const BFU& a,
                                            const BFU* __restrict__ bb) {
    #pragma unroll
    for (int j = 0; j < 8; ++j)
        acc[j] = __builtin_amdgcn_wmma_f32_16x16x32_bf16(
            false, a.v, false, bb[j].v, (short)0, acc[j], false, false);
}

static __device__ __forceinline__ int clampc(int c) {
    return c < NCHUNK ? c : (NCHUNK - 1);
}

// ------------- fused conv1d(GEMM) + bias + relu + layernorm + proj -------------
// Grid: 1024 blocks (B*N/16). Block: 256 threads = 8 waves (wave32).
__global__ __launch_bounds__(256) void conv_ln_proj_kernel(
    const unsigned* __restrict__ xb,      // bf16-pair x, padded (B,NP_,D/2) dwords
    const unsigned* __restrict__ bfrag,   // [48][1024][16] dwords
    const float* __restrict__ conv_b,
    const float* __restrict__ ln_g, const float* __restrict__ ln_b,
    const float* __restrict__ proj_w, const float* __restrict__ proj_b,
    float* __restrict__ probs_out,        // (B,N,2)
    float* __restrict__ vals_out)         // (B,N)
{
    __shared__ float hs[16][H_];          // 64 KB h tile

    const int tid  = threadIdx.x;
    const int wave = tid >> 5;
    const int lane = tid & 31;
    const int m    = lane & 15;
    const int half = lane >> 4;

    const int blk   = blockIdx.x;
    const int b     = blk >> 7;
    const int nbase = (blk & 127) << 4;
    const int obase = wave << 7;
    const int colbase = obase + m;

    const unsigned* abase =
        xb + ((size_t)b * NP_ + nbase + m) * ROWDW + half * 4;

    v8f acc[8];
    {
        v8f z = {0.f,0.f,0.f,0.f,0.f,0.f,0.f,0.f};
        #pragma unroll
        for (int j = 0; j < 8; ++j) acc[j] = z;
    }

    // Software-pipelined K loop, unroll 4:
    //   B fragments: ping-pong, prefetch distance 1 stage.
    //   A fragments: 4 rotating buffers, prefetch distance 2 stages
    //   (A-load issued 16 WMMAs + 32 B uint4-loads before first use -> no wait-0 stall).
    BFU aA, aB, aC, aD;
    BFU b0[8], b1[8];
    load_a(aA, abase, 0);
    load_a(aB, abase, 1);
    load_b(b0, bfrag, 0, colbase, half);

    #pragma unroll 1
    for (int c = 0; c < NCHUNK; c += 4) {
        load_b(b1, bfrag, c + 1, colbase, half);
        load_a(aC, abase, c + 2);
        mma8(acc, aA, b0);

        load_b(b0, bfrag, c + 2, colbase, half);
        load_a(aD, abase, c + 3);
        mma8(acc, aB, b1);

        load_b(b1, bfrag, c + 3, colbase, half);
        load_a(aA, abase, clampc(c + 4));
        mma8(acc, aC, b0);

        load_b(b0, bfrag, clampc(c + 4), colbase, half);
        load_a(aB, abase, clampc(c + 5));
        mma8(acc, aD, b1);
    }

    // ---- bias + relu, spill h tile to LDS ----
    #pragma unroll
    for (int j = 0; j < 8; ++j) {
        const int col  = obase + (j << 4) + m;
        const float bi = conv_b[col];
        #pragma unroll
        for (int v = 0; v < 8; ++v) {
            float val = acc[j][v] + bi;
            hs[half * 8 + v][col] = val > 0.f ? val : 0.f;
        }
    }
    __syncthreads();

    // ---- layernorm + projection: 16 threads per row, butterfly reduce ----
    const int r = tid >> 4;
    const int s = tid & 15;

    float sum = 0.f, sq = 0.f;
    for (int k = 0; k < 64; ++k) {
        float v = hs[r][s + (k << 4)];
        sum += v; sq += v * v;
    }
    #pragma unroll
    for (int mk = 1; mk <= 8; mk <<= 1) {
        sum += __shfl_xor(sum, mk, 32);
        sq  += __shfl_xor(sq,  mk, 32);
    }
    const float mu  = sum * (1.f / (float)H_);
    const float var = sq * (1.f / (float)H_) - mu * mu;
    const float rsg = rsqrtf(var + 1e-5f);

    float p0 = 0.f, p1 = 0.f;
    for (int k = 0; k < 64; ++k) {
        const int c = s + (k << 4);
        float hn = (hs[r][c] - mu) * rsg * ln_g[c] + ln_b[c];
        p0 += hn * proj_w[2 * c];
        p1 += hn * proj_w[2 * c + 1];
    }
    #pragma unroll
    for (int mk = 1; mk <= 8; mk <<= 1) {
        p0 += __shfl_xor(p0, mk, 32);
        p1 += __shfl_xor(p1, mk, 32);
    }

    if (s == 0) {
        const float q0 = p0 + proj_b[0];
        const float q1 = p1 + proj_b[1];
        const int   n  = nbase + r;
        const size_t off = ((size_t)b * N_ + n) * 2;
        probs_out[off]     = q0;
        probs_out[off + 1] = q1;
        const int g = (q1 > q0) ? 1 : 0;             // argmax, ties -> 0
        float val;
        if (n == N_ - 1)          val = (float)N_;   // vals[:, -1] = N (override)
        else if (n == 0 || g)     val = 0.f;         // zero_mask
        else                      val = (float)(n + 1);
        vals_out[(size_t)b * N_ + n] = val;
    }
}

// ------------- stable compaction of nonzero vals (= jnp.argsort trick) -------------
__global__ void compact_kernel(const float* __restrict__ vals,
                               float* __restrict__ groups) {
    const int b = blockIdx.x;
    if (threadIdx.x == 0) {
        int cnt = 0;
        for (int n = 0; n < N_; ++n) {
            float v = vals[b * N_ + n];
            if (v != 0.f) groups[b * N_ + cnt++] = v;
        }
        for (; cnt < N_; ++cnt) groups[b * N_ + cnt] = 0.f;
    }
}

// ------------- grouping matrix: one block per output row, float4 stores -------------
__global__ __launch_bounds__(256) void gm_kernel(const float* __restrict__ groups,
                                                 float* __restrict__ gm) {
    const int blk = blockIdx.x;
    const int b = blk >> 11;
    const int j = blk & (N_ - 1);
    const float gj   = groups[b * N_ + j];
    const float gjm1 = (j > 0) ? groups[b * N_ + j - 1] : 0.f;
    float4* out = (float4*)(gm + ((size_t)(b * N_ + j)) * N_);
    const int tid = threadIdx.x;
    #pragma unroll
    for (int i = 0; i < 2; ++i) {
        const int c4 = tid * 2 + i;                  // 512 float4 per row
        const int c0 = c4 * 4;
        float4 o;
        float* po = &o.x;
        #pragma unroll
        for (int q = 0; q < 4; ++q) {
            const float t  = (float)(c0 + q + 1);
            float hi = gj - t + 1.0f;
            hi = fminf(fmaxf(hi, 0.f), 1.f);
            float v;
            if (j == 0) {
                v = hi;
            } else {
                float lo = t - gjm1;
                lo = fminf(fmaxf(lo, 0.f), 1.f);
                v = hi * lo;
            }
            po[q] = v;
        }
        out[c4] = o;
    }
}

extern "C" void kernel_launch(void* const* d_in, const int* in_sizes, int n_in,
                              void* d_out, int out_size, void* d_ws, size_t ws_size,
                              hipStream_t stream) {
    (void)in_sizes; (void)n_in; (void)out_size; (void)ws_size;
    const float* x      = (const float*)d_in[0];
    const float* conv_w = (const float*)d_in[1];
    const float* conv_b = (const float*)d_in[2];
    const float* ln_g   = (const float*)d_in[3];
    const float* ln_b   = (const float*)d_in[4];
    const float* proj_w = (const float*)d_in[5];
    const float* proj_b = (const float*)d_in[6];
    float* out = (float*)d_out;

    // workspace layout (~21 MB total)
    char* ws = (char*)d_ws;
    unsigned* xb     = (unsigned*)(ws);                         // padded bf16 x: 16.02 MB
    unsigned* bfrag  = (unsigned*)(ws + (17u << 20));           // 3 MB
    float*    vals   = (float*)(ws + (20u << 20));              // 64 KB
    float*    groups = (float*)(ws + (20u << 20) + 65536);      // 64 KB

    cvt_x_kernel<<<16384, 256, 0, stream>>>(x, xb);
    zero_pad_kernel<<<B_, 256, 0, stream>>>(xb);
    cvt_w_kernel<<<3072, 256, 0, stream>>>(conv_w, bfrag);
    conv_ln_proj_kernel<<<(B_ * N_) / 16, 256, 0, stream>>>(
        xb, bfrag, conv_b, ln_g, ln_b, proj_w, proj_b, out, vals);
    compact_kernel<<<B_, 32, 0, stream>>>(vals, groups);
    gm_kernel<<<B_ * N_, 256, 0, stream>>>(groups, out + (size_t)B_ * N_ * 2);
}
